// GLABlock_62302795596602
// MI455X (gfx1250) — compile-verified, matrix operand order checked by reading, not run on hardware
//
#include <hip/hip_runtime.h>
#include <hip/hip_bf16.h>

// ---------------------------------------------------------------------------
// GLA block for MI455X (gfx1250, wave32, WMMA).
// GEMMs in bf16 WMMA (v_wmma_f32_16x16x32_bf16, f32 accumulate), 128x64 block
// tiles, double-buffered LDS, B packed K-major at weight-conversion time so
// both LDS tile stores are vectorized ds_store_b128.
// Scan is chunk-parallel (diagonal-decay affine recurrence).
// ---------------------------------------------------------------------------

typedef __bf16 bf16;
typedef __attribute__((ext_vector_type(16))) __bf16 v16bf;
typedef __attribute__((ext_vector_type(8)))  float  v8f;

#define D_MODEL 1024
#define N_HEADS 16
#define D_K     64
#define D_FF    2736
#define D_FF_P  2752   // padded to multiple of 32 (K) and 64 (tiles)
#define BT      2048   // B*T
#define T_LEN   1024
#define BH      32     // B * N_HEADS
#define N_CHUNK 16
#define L_CHUNK 64     // T_LEN / N_CHUNK

#define TILE_M 128
#define TILE_N 64
#define TILE_K 32

// ---------------------------------------------------------------------------
// RMSNorm: x[row, 1024] -> bf16 h = x * rsqrt(mean(x^2)+eps) * g
// ---------------------------------------------------------------------------
__global__ __launch_bounds__(256) void rmsnorm_bf16_kernel(
    const float* __restrict__ x, const float* __restrict__ g,
    bf16* __restrict__ out)
{
    __shared__ float red[256];
    const int row = blockIdx.x;
    const int tid = threadIdx.x;
    const float* xr = x + (size_t)row * D_MODEL;

    float vals[4];
    float s = 0.f;
#pragma unroll
    for (int i = 0; i < 4; ++i) {
        vals[i] = xr[tid + i * 256];
        s += vals[i] * vals[i];
    }
    red[tid] = s;
    __syncthreads();
#pragma unroll
    for (int off = 128; off > 0; off >>= 1) {
        if (tid < off) red[tid] += red[tid + off];
        __syncthreads();
    }
    const float r = rsqrtf(red[0] * (1.0f / D_MODEL) + 1e-6f);
#pragma unroll
    for (int i = 0; i < 4; ++i) {
        const int c = tid + i * 256;
        out[(size_t)row * D_MODEL + c] = (bf16)(vals[i] * r * g[c]);
    }
}

// ---------------------------------------------------------------------------
// Pack f32 weight [Ksrc, Nsrc] (row-major) into K-major bf16 panel:
//   dst[(rowOff + n) * ldK + k] = (k < Ksrc && n < Nsrc) ? src[k, n] : 0
// for n in [0, Npad), k in [0, Kdst). Output rows are contiguous in k so the
// GEMM can tile B with vectorized loads/stores (no in-loop transpose).
// ---------------------------------------------------------------------------
__global__ __launch_bounds__(256) void pack_T_kernel(
    bf16* __restrict__ dst, int ldK, int rowOff, int Kdst,
    const float* __restrict__ src, int Ksrc, int Nsrc, int Npad)
{
    const int i = blockIdx.x * 256 + threadIdx.x;
    const int total = Npad * Kdst;
    if (i >= total) return;
    const int n = i / Kdst;
    const int k = i - n * Kdst;
    float v = (k < Ksrc && n < Nsrc) ? src[(size_t)k * Nsrc + n] : 0.f;
    dst[(size_t)(rowOff + n) * ldK + k] = (bf16)v;
}

// ---------------------------------------------------------------------------
// bf16 WMMA GEMM: C[M,N](f32) = A[M,K](bf16,row) * B[K,N] (+ res), where B is
// supplied K-major (Bt[n][k], row stride K). Block tile 128x64, 256 threads
// (8 waves as 4x2 grid of 32x32 wave tiles). Each wave per K-step: 2 A-frags,
// 2 B-frags, 4 accumulators -> 4 v_wmma. Double-buffered LDS, one barrier per
// K-step. M%128==0, N%64==0, K%32==0 (guaranteed by padding).
// ---------------------------------------------------------------------------
__global__ __launch_bounds__(256) void gemm_bf16_wmma_kernel(
    const bf16* __restrict__ A, const bf16* __restrict__ Bt,
    float* __restrict__ C, const float* __restrict__ res,
    int M, int N, int K)
{
    __shared__ __align__(16) bf16 sA[2][TILE_M][TILE_K];   // [buf][m][k]
    __shared__ __align__(16) bf16 sBt[2][TILE_N][TILE_K];  // [buf][n][k]

    const int tid  = threadIdx.x;
    const int wave = tid >> 5;
    const int lane = tid & 31;
    const int l    = lane & 15;
    const int hi   = lane >> 4;

    const int m0 = blockIdx.y * TILE_M;
    const int n0 = blockIdx.x * TILE_N;

    const int waveM = wave >> 1;        // 0..3 -> 32-row slab
    const int waveN = wave & 1;         // 0..1 -> 32-col slab

    v8f acc[2][2] = {{{}, {}}, {{}, {}}};

    // Tile loader coords: A is 128x32 (2 x uint4/thread), Bt is 64x32
    // (1 x uint4/thread) -- same (row, k-offset) pattern for both.
    const int tr = tid >> 2;            // 0..63
    const int tc = (tid & 3) * 8;

    const int nk = K / TILE_K;

    // --- preload tile 0 into buffer 0 ---
    {
        *(uint4*)&sA[0][tr][tc]      = *(const uint4*)(A  + (size_t)(m0 + tr) * K + tc);
        *(uint4*)&sA[0][tr + 64][tc] = *(const uint4*)(A  + (size_t)(m0 + tr + 64) * K + tc);
        *(uint4*)&sBt[0][tr][tc]     = *(const uint4*)(Bt + (size_t)(n0 + tr) * K + tc);
    }
    __syncthreads();

    for (int kt = 0; kt < nk; ++kt) {
        const int cur = kt & 1;
        const int nxt = cur ^ 1;
        const bool hasNext = (kt + 1) < nk;

        // --- issue next-tile global loads early (latency hidden by WMMAs) ---
        uint4 a0r, a1r, b0r;
        if (hasNext) {
            const int k1 = (kt + 1) * TILE_K;
            a0r = *(const uint4*)(A  + (size_t)(m0 + tr) * K + k1 + tc);
            a1r = *(const uint4*)(A  + (size_t)(m0 + tr + 64) * K + k1 + tc);
            b0r = *(const uint4*)(Bt + (size_t)(n0 + tr) * K + k1 + tc);
            if (kt + 2 < nk) {  // global_prefetch_b8 for tile kt+2
                const int k2 = (kt + 2) * TILE_K;
                __builtin_prefetch(A  + (size_t)(m0 + tr) * K + k2 + tc, 0, 1);
                __builtin_prefetch(Bt + (size_t)(n0 + tr) * K + k2 + tc, 0, 1);
            }
        }

        // --- build fragments from current buffer ---
        // 16-bit A layout: lanes 0-15 hold K{0..7,16..23}, lanes 16-31
        // K{8..15,24..31}, two packed elements per VGPR.
        v16bf af0, af1, bf0, bf1;
        {
            const int mA = waveM * 32 + l;
            const int nA = waveN * 32 + l;
#pragma unroll
            for (int p = 0; p < 8; ++p) {
                const int e  = 2 * p;
                const int kk = ((e < 8) ? e : e + 8) + hi * 8;
                union { unsigned u; bf16 h[2]; } cv;
                cv.u = *(const unsigned*)&sA[cur][mA][kk];
                af0[e] = cv.h[0]; af0[e + 1] = cv.h[1];
                cv.u = *(const unsigned*)&sA[cur][mA + 16][kk];
                af1[e] = cv.h[0]; af1[e + 1] = cv.h[1];
                cv.u = *(const unsigned*)&sBt[cur][nA][kk];
                bf0[e] = cv.h[0]; bf0[e + 1] = cv.h[1];
                cv.u = *(const unsigned*)&sBt[cur][nA + 16][kk];
                bf1[e] = cv.h[0]; bf1[e + 1] = cv.h[1];
            }
        }

        acc[0][0] = __builtin_amdgcn_wmma_f32_16x16x32_bf16(false, af0, false, bf0, (short)0, acc[0][0], false, false);
        acc[0][1] = __builtin_amdgcn_wmma_f32_16x16x32_bf16(false, af0, false, bf1, (short)0, acc[0][1], false, false);
        acc[1][0] = __builtin_amdgcn_wmma_f32_16x16x32_bf16(false, af1, false, bf0, (short)0, acc[1][0], false, false);
        acc[1][1] = __builtin_amdgcn_wmma_f32_16x16x32_bf16(false, af1, false, bf1, (short)0, acc[1][1], false, false);

        // --- stash next tile into the other buffer (vectorized) ---
        if (hasNext) {
            *(uint4*)&sA[nxt][tr][tc]      = a0r;
            *(uint4*)&sA[nxt][tr + 64][tc] = a1r;
            *(uint4*)&sBt[nxt][tr][tc]     = b0r;
        }
        __syncthreads();
    }

    // --- store: C/D layout: lane<16 -> M=r, lane>=16 -> M=r+8; N=l ---
    const int rowBase = m0 + waveM * 32 + hi * 8;
    const int colBase = n0 + waveN * 32 + l;
    if (res) {
#pragma unroll
        for (int mi = 0; mi < 2; ++mi)
#pragma unroll
            for (int ni = 0; ni < 2; ++ni)
#pragma unroll
                for (int r = 0; r < 8; ++r) {
                    const size_t idx = (size_t)(rowBase + mi * 16 + r) * N
                                     + colBase + ni * 16;
                    C[idx] = acc[mi][ni][r] + res[idx];
                }
    } else {
#pragma unroll
        for (int mi = 0; mi < 2; ++mi)
#pragma unroll
            for (int ni = 0; ni < 2; ++ni)
#pragma unroll
                for (int r = 0; r < 8; ++r) {
                    const size_t idx = (size_t)(rowBase + mi * 16 + r) * N
                                     + colBase + ni * 16;
                    C[idx] = acc[mi][ni][r];
                }
    }
}

// ---------------------------------------------------------------------------
// Post-process fused qkvg GEMM output [BT, 4096]:
//   q,k -> l2norm per (token, head); g -> sigmoid; v -> copy.
// Writes scan layout [B*H, T, 64] (f32). Grid: B*T*H blocks of 64 threads.
// ---------------------------------------------------------------------------
__global__ __launch_bounds__(64) void postproc_qkvg_kernel(
    const float* __restrict__ qkvg,
    float* __restrict__ qs, float* __restrict__ ks,
    float* __restrict__ as, float* __restrict__ vs)
{
    __shared__ float red[64];
    const int blk = blockIdx.x;
    const int h   = blk & (N_HEADS - 1);
    const int bt  = blk >> 4;
    const int b   = bt >> 10;
    const int t   = bt & (T_LEN - 1);
    const int tid = threadIdx.x;

    const float* rowp = qkvg + (size_t)bt * (4 * D_MODEL);
    const float qv = rowp[              h * D_K + tid];
    const float kv = rowp[D_MODEL     + h * D_K + tid];
    const float vv = rowp[2 * D_MODEL + h * D_K + tid];
    const float gv = rowp[3 * D_MODEL + h * D_K + tid];

    red[tid] = qv * qv;
    __syncthreads();
#pragma unroll
    for (int off = 32; off > 0; off >>= 1) {
        if (tid < off) red[tid] += red[tid + off];
        __syncthreads();
    }
    const float qn = sqrtf(red[0]);
    __syncthreads();

    red[tid] = kv * kv;
    __syncthreads();
#pragma unroll
    for (int off = 32; off > 0; off >>= 1) {
        if (tid < off) red[tid] += red[tid + off];
        __syncthreads();
    }
    const float kn = sqrtf(red[0]);

    const size_t o = ((size_t)(b * N_HEADS + h) * T_LEN + t) * D_K + tid;
    qs[o] = qv / fmaxf(qn, 1e-12f);
    ks[o] = kv / fmaxf(kn, 1e-12f);
    as[o] = 1.f / (1.f + __expf(-gv));
    vs[o] = vv;
}

// ---------------------------------------------------------------------------
// Chunk-parallel GLA scan. Decay is diagonal in k, so the recurrence is an
// elementwise affine map and composes chunk-wise:
//   pass1: per (bh, chunk): L_c (state from zero) and P_c[k] = prod alpha.
//   pass2: per bh: S_init_{c+1} = P_c (*) S_init_c + L_c  (16 serial steps).
//   pass3: per (bh, chunk): replay chunk from S_init_c, emit o (bf16).
// One thread owns one v-column of the 64x64 state (64 VGPRs); a/k/q broadcast
// through LDS each step -> no cross-lane reductions.
// ---------------------------------------------------------------------------
__global__ __launch_bounds__(64) void gla_scan_pass1_kernel(
    const float* __restrict__ ks, const float* __restrict__ as,
    const float* __restrict__ vs,
    float* __restrict__ Lbuf, float* __restrict__ Pbuf)
{
    __shared__ float sa[64], sk[64];
    const int bc  = blockIdx.x;                 // bh * N_CHUNK + c
    const int tid = threadIdx.x;                // v-column
    const size_t base = (size_t)bc * L_CHUNK * D_K;

    float L[64];
#pragma unroll
    for (int i = 0; i < 64; ++i) L[i] = 0.f;
    float P = 1.f;

    for (int tc = 0; tc < L_CHUNK; ++tc) {
        const size_t off = base + (size_t)tc * D_K + tid;
        const float av = as[off];
        const float kv = ks[off];
        const float vv = vs[off];
        __syncthreads();
        sa[tid] = av; sk[tid] = kv;
        __syncthreads();
        P *= av;                               // tid doubles as the k index
#pragma unroll
        for (int kk = 0; kk < 64; ++kk)
            L[kk] = sa[kk] * L[kk] + sk[kk] * vv;
    }

#pragma unroll
    for (int kk = 0; kk < 64; ++kk)
        Lbuf[((size_t)bc * D_K + kk) * D_K + tid] = L[kk];
    Pbuf[(size_t)bc * D_K + tid] = P;
}

__global__ __launch_bounds__(64) void gla_scan_pass2_kernel(
    const float* __restrict__ Lbuf, const float* __restrict__ Pbuf,
    float* __restrict__ Sinit)
{
    __shared__ float sp[64];
    const int bh  = blockIdx.x;
    const int tid = threadIdx.x;

    float S[64];
#pragma unroll
    for (int i = 0; i < 64; ++i) S[i] = 0.f;

    for (int c = 0; c < N_CHUNK; ++c) {
        const size_t bc = (size_t)bh * N_CHUNK + c;
#pragma unroll
        for (int kk = 0; kk < 64; ++kk)
            Sinit[(bc * D_K + kk) * D_K + tid] = S[kk];
        __syncthreads();
        sp[tid] = Pbuf[bc * D_K + tid];
        __syncthreads();
#pragma unroll
        for (int kk = 0; kk < 64; ++kk)
            S[kk] = sp[kk] * S[kk] + Lbuf[(bc * D_K + kk) * D_K + tid];
    }
}

__global__ __launch_bounds__(64) void gla_scan_pass3_kernel(
    const float* __restrict__ qs, const float* __restrict__ ks,
    const float* __restrict__ as, const float* __restrict__ vs,
    const float* __restrict__ Sinit, bf16* __restrict__ o_bf)
{
    __shared__ float sa[64], sk[64], sq[64];
    const int bc  = blockIdx.x;
    const int bh  = bc >> 4;                   // / N_CHUNK
    const int c   = bc & (N_CHUNK - 1);
    const int b   = bh >> 4;                   // / N_HEADS
    const int h   = bh & (N_HEADS - 1);
    const int tid = threadIdx.x;

    float S[64];
#pragma unroll
    for (int kk = 0; kk < 64; ++kk)
        S[kk] = Sinit[((size_t)bc * D_K + kk) * D_K + tid];

    const size_t base = (size_t)bc * L_CHUNK * D_K;
    for (int tc = 0; tc < L_CHUNK; ++tc) {
        const size_t off = base + (size_t)tc * D_K + tid;
        const float av = as[off];
        const float kv = ks[off];
        const float qv = qs[off];
        const float vv = vs[off];
        __syncthreads();
        sa[tid] = av; sk[tid] = kv; sq[tid] = qv;
        __syncthreads();

        float acc = 0.f;
#pragma unroll
        for (int kk = 0; kk < 64; ++kk) {
            S[kk] = sa[kk] * S[kk] + sk[kk] * vv;
            acc += sq[kk] * S[kk];
        }
        const int t = c * L_CHUNK + tc;
        o_bf[(size_t)(b * T_LEN + t) * D_MODEL + h * D_K + tid] = (bf16)acc;
    }
}

// ---------------------------------------------------------------------------
// SwiGLU: u[row, j] = silu(ff[row, j]) * ff[row, 2752 + j] for j < 2736,
// zero in padding columns. Output bf16 [BT, 2752].
// ---------------------------------------------------------------------------
__global__ __launch_bounds__(256) void swiglu_kernel(
    const float* __restrict__ ff, bf16* __restrict__ u)
{
    const int i = blockIdx.x * 256 + threadIdx.x;
    if (i >= BT * D_FF_P) return;
    const int row = i / D_FF_P;
    const int col = i - row * D_FF_P;
    float val = 0.f;
    if (col < D_FF) {
        const float g1 = ff[(size_t)row * (2 * D_FF_P) + col];
        const float g2 = ff[(size_t)row * (2 * D_FF_P) + D_FF_P + col];
        val = (g1 / (1.f + __expf(-g1))) * g2;
    }
    u[i] = (bf16)val;
}

// ---------------------------------------------------------------------------
// Launch
// ---------------------------------------------------------------------------
extern "C" void kernel_launch(void* const* d_in, const int* in_sizes, int n_in,
                              void* d_out, int out_size, void* d_ws, size_t ws_size,
                              hipStream_t stream)
{
    const float* x      = (const float*)d_in[0];
    const float* Wq     = (const float*)d_in[1];
    const float* Wk     = (const float*)d_in[2];
    const float* Wv     = (const float*)d_in[3];
    const float* Wg     = (const float*)d_in[4];
    const float* Wo     = (const float*)d_in[5];
    const float* w1     = (const float*)d_in[6];
    const float* w2     = (const float*)d_in[7];
    const float* w3     = (const float*)d_in[8];
    const float* g_attn = (const float*)d_in[9];
    const float* g_ffn  = (const float*)d_in[10];
    float* out = (float*)d_out;

    char* ws = (char*)d_ws;
    const size_t MiB = 1024 * 1024;
    // Workspace regions (sequentially reused across pipeline phases):
    bf16*  hb    = (bf16*)(ws + 0);             //  4 MiB: bf16 activations (h, then h2)
    bf16*  wpack = (bf16*)(ws + 4   * MiB);     // 12 MiB: packed K-major bf16 weights
    float* gout  = (float*)(ws + 16  * MiB);    // 48 MiB: f32 GEMM out (qkvg, then ffn)
    float* qs    = (float*)(ws + 64  * MiB);    //  8 MiB each: scan inputs
    float* ks    = (float*)(ws + 72  * MiB);
    float* as    = (float*)(ws + 80  * MiB);
    float* vs    = (float*)(ws + 88  * MiB);
    bf16*  obf   = (bf16*)(ws + 96  * MiB);     // 12 MiB: o (bf16), later u (bf16)
    bf16*  ubf   = (bf16*)(ws + 96  * MiB);
    float* x2    = (float*)(ws + 108 * MiB);    //  8 MiB: residual after attention
    float* Lbuf  = (float*)(ws + 116 * MiB);    //  8 MiB: chunk local states
    float* Sinit = (float*)(ws + 124 * MiB);    //  8 MiB: chunk initial states
    float* Pbuf  = (float*)(ws + 132 * MiB);    // 128 KiB: chunk decay products

    const dim3 blk256(256), blk64(64);

    // ---- 1. h = rmsnorm(x, g_attn) (bf16) ----
    rmsnorm_bf16_kernel<<<BT, blk256, 0, stream>>>(x, g_attn, hb);

    // ---- 2. pack [Wq|Wk|Wv|Wg] -> K-major bf16 Bt[4096][1024] ----
    {
        const int thr = D_MODEL * D_MODEL;
        const int g = (thr + 255) / 256;
        pack_T_kernel<<<g, blk256, 0, stream>>>(wpack, D_MODEL,    0, D_MODEL, Wq, D_MODEL, D_MODEL, D_MODEL);
        pack_T_kernel<<<g, blk256, 0, stream>>>(wpack, D_MODEL, 1024, D_MODEL, Wk, D_MODEL, D_MODEL, D_MODEL);
        pack_T_kernel<<<g, blk256, 0, stream>>>(wpack, D_MODEL, 2048, D_MODEL, Wv, D_MODEL, D_MODEL, D_MODEL);
        pack_T_kernel<<<g, blk256, 0, stream>>>(wpack, D_MODEL, 3072, D_MODEL, Wg, D_MODEL, D_MODEL, D_MODEL);
    }

    // ---- 3. fused qkvg GEMM: [2048,1024] x [1024,4096] -> f32 ----
    gemm_bf16_wmma_kernel<<<dim3(4096 / TILE_N, BT / TILE_M), blk256, 0, stream>>>(
        hb, wpack, gout, nullptr, BT, 4096, D_MODEL);

    // ---- 4. l2norm(q,k), sigmoid(g), reorder for scan ----
    postproc_qkvg_kernel<<<BT * N_HEADS, blk64, 0, stream>>>(gout, qs, ks, as, vs);

    // ---- 5. pack Wo -> K-major bf16 Bt[1024][1024] (wpack free again) ----
    {
        const int thr = D_MODEL * D_MODEL;
        pack_T_kernel<<<(thr + 255) / 256, blk256, 0, stream>>>(
            wpack, D_MODEL, 0, D_MODEL, Wo, D_MODEL, D_MODEL, D_MODEL);
    }

    // ---- 6. chunk-parallel GLA scan -> o (bf16) ----
    gla_scan_pass1_kernel<<<BH * N_CHUNK, blk64, 0, stream>>>(ks, as, vs, Lbuf, Pbuf);
    gla_scan_pass2_kernel<<<BH, blk64, 0, stream>>>(Lbuf, Pbuf, Sinit);
    gla_scan_pass3_kernel<<<BH * N_CHUNK, blk64, 0, stream>>>(qs, ks, as, vs, Sinit, obf);

    // ---- 7. x2 = x + o @ Wo ----
    gemm_bf16_wmma_kernel<<<dim3(D_MODEL / TILE_N, BT / TILE_M), blk256, 0, stream>>>(
        obf, wpack, x2, x, BT, D_MODEL, D_MODEL);

    // ---- 8. h2 = rmsnorm(x2, g_ffn) (reuse hb) ----
    rmsnorm_bf16_kernel<<<BT, blk256, 0, stream>>>(x2, g_ffn, hb);

    // ---- 9. pack [w1|w2] -> K-major bf16 Bt[5504][1024] (2736->2752 pad) ----
    {
        const int thr = D_FF_P * D_MODEL;
        const int g = (thr + 255) / 256;
        pack_T_kernel<<<g, blk256, 0, stream>>>(wpack, D_MODEL,      0, D_MODEL, w1, D_MODEL, D_FF, D_FF_P);
        pack_T_kernel<<<g, blk256, 0, stream>>>(wpack, D_MODEL, D_FF_P, D_MODEL, w2, D_MODEL, D_FF, D_FF_P);
    }

    // ---- 10. fused ffn GEMM: [2048,1024] x [1024,5504] -> f32 ----
    gemm_bf16_wmma_kernel<<<dim3((2 * D_FF_P) / TILE_N, BT / TILE_M), blk256, 0, stream>>>(
        hb, wpack, gout, nullptr, BT, 2 * D_FF_P, D_MODEL);

    // ---- 11. u = silu(h@w1) * (h@w2) (bf16, padded K) ----
    swiglu_kernel<<<(BT * D_FF_P + 255) / 256, blk256, 0, stream>>>(gout, ubf);

    // ---- 12. pack w3 -> K-major bf16 Bt[1024][2752] (zero pad k>=2736) ----
    {
        const int thr = D_MODEL * D_FF_P;
        pack_T_kernel<<<(thr + 255) / 256, blk256, 0, stream>>>(
            wpack, D_FF_P, 0, D_FF_P, w3, D_FF, D_MODEL, D_MODEL);
    }

    // ---- 13. out = x2 + u @ w3 ----
    gemm_bf16_wmma_kernel<<<dim3(D_MODEL / TILE_N, BT / TILE_M), blk256, 0, stream>>>(
        ubf, wpack, out, x2, BT, D_MODEL, D_FF_P);
}
